// CppnPotentialCA_10917806867137
// MI455X (gfx1250) — matrix operand order repaired and checked by reading, not compiled, not run on hardware
//
#include <hip/hip_runtime.h>

// ---------------------------------------------------------------------------
// CppnPotentialCA on MI455X (gfx1250, wave32, WMMA)
//
// pot_e = depthwise 9x9x9 wrap-conv of channel c0[e]  -> implicit GEMM:
//   group e's by c0 (device counting sort), A = weights (16 rows x 736 K),
//   B = im2col of the channel (736 x 16 x-positions), D = pot (e x x).
// Split precision: w = whi+wlo, x = xhi+xlo (f16); f16*f16 is exact in the
// f32 WMMA accumulator, so whi*xhi + whi*xlo + wlo*xhi ~ fp32 accuracy.
// Then growth = exp(-(pot-m)^2/(2 s^2))*2-1, segment-summed into LDS field
// by c1[e] (ds_add_f32), fused final: out = clip(x + field/T, 0, 10).
//
// B-gather trick: tap k (k<736) decodes to LDS word addr = k + col + 127*r
// with r = k/9 = (k*3641)>>15 (exact for k<=735). Taps 729..735 land in a
// zeroed 144-word pad after the halo, so no predication is needed.
// ---------------------------------------------------------------------------

typedef __attribute__((ext_vector_type(16))) _Float16     v16h;
typedef __attribute__((ext_vector_type(8)))  float        v8f;
typedef __attribute__((ext_vector_type(8)))  unsigned int v8u;

#define TAPS 729
#define NKK  23                 // ceil(736/32) K-steps of 32
#define STAGE_WORDS (81*136)
#define STAGE_PAD   144
#define WS_NSLOTS 0
#define WS_INVT   4
#define WS_ME     16
#define WS_INV2S2 272
#define WS_C1     528
#define WS_SLOTCH 784
#define WS_ROWE   1040          // int rowE[64*16]
#define WS_AFRAG  8192          // uint Afrag[slot][kk][part][lane*8+v]
#define AFRAG_SLOT_STRIDE (NKK*512)   // words per slot (hi+lo, 256 each)
#define WS_XT (4u*1024u*1024u)        // channel-major transpose (128 MiB)

// ---------------------------------------------------------------------------
// Prep: counting-sort e by c0, build slot table, params, and pre-swizzled
// A fragments (f16 hi/lo) in the documented 16-bit A-matrix lane/K layout.
// ---------------------------------------------------------------------------
__global__ void __launch_bounds__(256) prep_kernel(
    const float* __restrict__ kern, const float* __restrict__ m,
    const float* __restrict__ s, const float* __restrict__ T,
    const int* __restrict__ c0, const int* __restrict__ c1,
    unsigned char* __restrict__ ws)
{
    __shared__ int cnt[16], off[16], cur[16], perm[64];
    __shared__ int slotCh_s[64], slotRB_s[64];
    __shared__ short rowE_s[64 * 16];
    __shared__ int ns_s;
    int t = threadIdx.x;

    if (t < 16) cnt[t] = 0;
    __syncthreads();
    if (t < 64) atomicAdd(&cnt[c0[t]], 1);
    __syncthreads();
    if (t == 0) {
        int a = 0;
        for (int c = 0; c < 16; ++c) { off[c] = a; cur[c] = a; a += cnt[c]; }
    }
    __syncthreads();
    if (t < 64) { int p = atomicAdd(&cur[c0[t]], 1); perm[p] = t; }
    __syncthreads();
    if (t == 0) {
        int ns = 0;
        for (int c = 0; c < 16; ++c)
            for (int rb = 0; rb < cnt[c]; rb += 16) {
                slotCh_s[ns] = c; slotRB_s[ns] = off[c] + rb; ++ns;
            }
        ns_s = ns;
        *(int*)(ws + WS_NSLOTS) = ns;
        *(float*)(ws + WS_INVT) = 1.0f / T[0];
    }
    __syncthreads();
    int ns = ns_s;
    for (int i = t; i < ns * 16; i += 256) {
        int slot = i >> 4, r = i & 15;
        int c = slotCh_s[slot];
        int idx = slotRB_s[slot] + r;
        int e = (idx < off[c] + cnt[c]) ? perm[idx] : -1;
        rowE_s[i] = (short)e;
        ((int*)(ws + WS_ROWE))[i] = e;
    }
    if (t < 64) {
        ((float*)(ws + WS_ME))[t] = m[t];
        float sv = s[t];
        ((float*)(ws + WS_INV2S2))[t] = 1.0f / (2.0f * sv * sv);
        ((int*)(ws + WS_C1))[t] = c1[t];
    }
    if (t < 64 && t < ns) ((int*)(ws + WS_SLOTCH))[t] = slotCh_s[t];
    __syncthreads();

    // A fragments. 16-bit A 16x32 layout (ISA 7.12.2):
    // lane L (M = L%16), VGPR v, half h -> K = (v<4 ? 2v+h : 16+2(v-4)+h) + 8*(L>=16)
    unsigned int* af = (unsigned int*)(ws + WS_AFRAG);
    int total = ns * NKK * 512;
    for (int w = t; w < total; w += 256) {
        int v = w & 7, L = (w >> 3) & 31, part = (w >> 8) & 1;
        int rest = w >> 9, kk = rest % NKK, slot = rest / NKK;
        int hiL = L >> 4, r = L & 15;
        int e = rowE_s[slot * 16 + r];
        unsigned int word = 0;
        if (e >= 0) {
            #pragma unroll
            for (int half = 0; half < 2; ++half) {
                int koff = (v < 4 ? 2 * v + half : 16 + 2 * (v - 4) + half) + 8 * hiL;
                int k = kk * 32 + koff;
                float wv = (k < TAPS) ? kern[e * TAPS + k] : 0.0f;
                _Float16 hi = (_Float16)wv;
                _Float16 val = part ? (_Float16)(wv - (float)hi) : hi;
                unsigned short b = __builtin_bit_cast(unsigned short, val);
                word |= ((unsigned int)b) << (16 * half);
            }
        }
        af[w] = word;
    }
}

// ---------------------------------------------------------------------------
// Transpose x (z,y,x,c) -> xt (c,z,y,x) so per-channel halo staging is
// contiguous (full 64B-line utilization out of L2).
// ---------------------------------------------------------------------------
__global__ void __launch_bounds__(128) transpose_kernel(
    const float* __restrict__ x, float* __restrict__ xt)
{
    __shared__ float tile[16][132];
    int y = blockIdx.x, z = blockIdx.y;
    int t = threadIdx.x;  // 128 threads = 128 x-positions
    const float4* xp = (const float4*)(x + (((size_t)z * 128 + y) * 128) * 16);
    #pragma unroll
    for (int c4 = 0; c4 < 4; ++c4) {
        float4 v = xp[t * 4 + c4];
        tile[c4 * 4 + 0][t] = v.x;
        tile[c4 * 4 + 1][t] = v.y;
        tile[c4 * 4 + 2][t] = v.z;
        tile[c4 * 4 + 3][t] = v.w;
    }
    __syncthreads();
    int c = t >> 3, xs = (t & 7) * 16;
    float* dst = xt + (((size_t)c * 128 + z) * 128 + y) * 128;
    #pragma unroll
    for (int j = 0; j < 4; ++j) {
        float4 v = { tile[c][xs + 4 * j], tile[c][xs + 4 * j + 1],
                     tile[c][xs + 4 * j + 2], tile[c][xs + 4 * j + 3] };
        *(float4*)(dst + xs + 4 * j) = v;
    }
}

// ---------------------------------------------------------------------------
// Main fused kernel: one WG per (z,y) line, 4 waves x 2 x-tiles of 16.
// ---------------------------------------------------------------------------
__global__ void __launch_bounds__(128) ca_kernel(
    const float* __restrict__ x, const float* __restrict__ xt,
    const unsigned char* __restrict__ ws, float* __restrict__ out, int useXT)
{
    __shared__ unsigned int stage[STAGE_WORDS + STAGE_PAD]; // halo + zero pad
    __shared__ float field[16 * 128];                       // segment-sum accumulator
    __shared__ float rowM[16], rowI[16];
    __shared__ int rowC1[16], rowEl[16];

    const int   nslots = *(const int*)(ws + WS_NSLOTS);
    const float invT   = *(const float*)(ws + WS_INVT);
    const float* me    = (const float*)(ws + WS_ME);
    const float* i2s   = (const float*)(ws + WS_INV2S2);
    const int*  c1p    = (const int*)(ws + WS_C1);
    const int*  slotCh = (const int*)(ws + WS_SLOTCH);
    const int*  rowE   = (const int*)(ws + WS_ROWE);
    const unsigned int* afrag = (const unsigned int*)(ws + WS_AFRAG);

    int y = blockIdx.x, z = blockIdx.y;
    int t = threadIdx.x;
    int lane = t & 31, wave = t >> 5;
    int n = lane & 15, hiL = lane >> 4;
    int col0 = wave * 16 + n;          // x-tile of q=0
    int col1 = col0 + 64;              // x-tile of q=1

    for (int i = t; i < 16 * 128; i += 128) field[i] = 0.0f;
    for (int i = t; i < STAGE_PAD; i += 128) stage[STAGE_WORDS + i] = 0;

    for (int slot = 0; slot < nslots; ++slot) {
        int ch = slotCh[slot];
        __syncthreads();
        // Stage wrap halo of channel ch, split into f16 hi/lo per word.
        if (useXT) {
            for (int i = t; i < STAGE_WORDS; i += 128) {
                int row = i / 136, xx = i - row * 136;
                int dz = row / 9, dy = row - dz * 9;
                int zz = (z + dz - 4) & 127, yy = (y + dy - 4) & 127, xm = (xx - 4) & 127;
                float v = xt[(((size_t)ch * 128 + zz) * 128 + yy) * 128 + xm];
                _Float16 h = (_Float16)v;
                _Float16 lo = (_Float16)(v - (float)h);
                stage[i] = (unsigned int)__builtin_bit_cast(unsigned short, h)
                         | ((unsigned int)__builtin_bit_cast(unsigned short, lo) << 16);
            }
        } else {
            for (int i = t; i < STAGE_WORDS; i += 128) {
                int row = i / 136, xx = i - row * 136;
                int dz = row / 9, dy = row - dz * 9;
                int zz = (z + dz - 4) & 127, yy = (y + dy - 4) & 127, xm = (xx - 4) & 127;
                float v = x[((((size_t)zz * 128 + yy) * 128 + xm) << 4) + ch];
                _Float16 h = (_Float16)v;
                _Float16 lo = (_Float16)(v - (float)h);
                stage[i] = (unsigned int)__builtin_bit_cast(unsigned short, h)
                         | ((unsigned int)__builtin_bit_cast(unsigned short, lo) << 16);
            }
        }
        if (t < 16) {
            int e = rowE[slot * 16 + t];
            rowEl[t] = e;
            rowM[t]  = (e >= 0) ? me[e]  : 0.0f;
            rowI[t]  = (e >= 0) ? i2s[e] : 0.0f;
            rowC1[t] = (e >= 0) ? c1p[e] : 0;
        }
        __syncthreads();

        v8f acc0 = {0.f, 0.f, 0.f, 0.f, 0.f, 0.f, 0.f, 0.f};
        v8f acc1 = acc0;

        const unsigned int* ab = afrag + (size_t)slot * AFRAG_SLOT_STRIDE;
        for (int kk = 0; kk < NKK; ++kk) {
            const unsigned int* abk = ab + kk * 512;
            v8u ahw = *(const v8u*)(abk + lane * 8);          // 2x global_load_b128
            v8u alw = *(const v8u*)(abk + 256 + lane * 8);
            v16h ah = __builtin_bit_cast(v16h, ahw);
            v16h al = __builtin_bit_cast(v16h, alw);
            if (kk + 1 < NKK) __builtin_prefetch(abk + 512 + lane * 8, 0, 0);

            int kbase = kk * 32 + hiL * 8;   // B 32x16 f16: elem i -> K = 8*hi + i (i<8), 16 + 8*hi + (i-8) (i>=8)
            #pragma unroll
            for (int q = 0; q < 2; ++q) {
                int base = kbase + (q ? col1 : col0);   // addr = k + col + 127*(k/9)
                unsigned int wv[16];
                #pragma unroll
                for (int i = 0; i < 16; ++i) {
                    int k = kbase + i + (i >= 8 ? 8 : 0);
                    int r = (k * 3641) >> 15;           // k/9, exact for k<=735
                    wv[i] = stage[base + 127 * r + i + (i >= 8 ? 8 : 0)];  // taps>=729 hit zero pad
                }
                v8u bhw, blw;
                #pragma unroll
                for (int j = 0; j < 8; ++j) {
                    bhw[j] = __builtin_amdgcn_perm(wv[2 * j + 1], wv[2 * j], 0x05040100u);
                    blw[j] = __builtin_amdgcn_perm(wv[2 * j + 1], wv[2 * j], 0x07060302u);
                }
                v16h bh = __builtin_bit_cast(v16h, bhw);
                v16h bl = __builtin_bit_cast(v16h, blw);
                v8f a = q ? acc1 : acc0;
                a = __builtin_amdgcn_wmma_f32_16x16x32_f16(false, ah, false, bh, (short)0, a, false, false);
                a = __builtin_amdgcn_wmma_f32_16x16x32_f16(false, ah, false, bl, (short)0, a, false, false);
                a = __builtin_amdgcn_wmma_f32_16x16x32_f16(false, al, false, bh, (short)0, a, false, false);
                if (q) acc1 = a; else acc0 = a;
            }
        }

        // Growth + segment-sum (D layout: VGPR r -> M = r + 8*(lane>=16), col = lane%16)
        #pragma unroll
        for (int r = 0; r < 8; ++r) {
            int M = r + 8 * hiL;
            int e = rowEl[M];
            if (e >= 0) {
                float mM = rowM[M], iM = rowI[M];
                int fb = rowC1[M] * 128;
                float d0 = acc0[r] - mM;
                float g0 = __expf(-d0 * d0 * iM) * 2.0f - 1.0f;
                atomicAdd(&field[fb + col0], g0);               // ds_add_f32
                float d1 = acc1[r] - mM;
                float g1 = __expf(-d1 * d1 * iM) * 2.0f - 1.0f;
                atomicAdd(&field[fb + col1], g1);
            }
        }
    }
    __syncthreads();

    // Fused finalize: out = clip(x + field/T, 0, 10)
    const float4* xin = (const float4*)x;
    float4* o = (float4*)out;
    size_t base4 = ((size_t)z * 128 + y) * 128 * 4;
    for (int i = t; i < 512; i += 128) {
        int xg = i >> 2, c4 = i & 3;
        size_t gi = base4 + (size_t)xg * 4 + c4;
        float4 xv = xin[gi];
        float4 r;
        r.x = fminf(fmaxf(xv.x + field[(c4 * 4 + 0) * 128 + xg] * invT, 0.0f), 10.0f);
        r.y = fminf(fmaxf(xv.y + field[(c4 * 4 + 1) * 128 + xg] * invT, 0.0f), 10.0f);
        r.z = fminf(fmaxf(xv.z + field[(c4 * 4 + 2) * 128 + xg] * invT, 0.0f), 10.0f);
        r.w = fminf(fmaxf(xv.w + field[(c4 * 4 + 3) * 128 + xg] * invT, 0.0f), 10.0f);
        o[gi] = r;
    }
}

// ---------------------------------------------------------------------------
extern "C" void kernel_launch(void* const* d_in, const int* in_sizes, int n_in,
                              void* d_out, int out_size, void* d_ws, size_t ws_size,
                              hipStream_t stream)
{
    const float* x    = (const float*)d_in[0];
    const float* kern = (const float*)d_in[1];
    const float* m    = (const float*)d_in[2];
    const float* s    = (const float*)d_in[3];
    const float* T    = (const float*)d_in[4];
    const int*   c0   = (const int*)d_in[5];
    const int*   c1   = (const int*)d_in[6];
    float* out = (float*)d_out;
    unsigned char* ws = (unsigned char*)d_ws;

    prep_kernel<<<1, 256, 0, stream>>>(kern, m, s, T, c0, c1, ws);

    size_t xt_need = (size_t)WS_XT + (size_t)16 * 128 * 128 * 128 * 4;
    int useXT = (ws_size >= xt_need) ? 1 : 0;      // uniform, capture-safe
    float* xtp = (float*)(ws + WS_XT);
    if (useXT) transpose_kernel<<<dim3(128, 128), 128, 0, stream>>>(x, xtp);

    ca_kernel<<<dim3(128, 128), 128, 0, stream>>>(x, xtp, ws, out, useXT);
}
